// CellGraphSAGE_17635135717841
// MI455X (gfx1250) — matrix operand order repaired and checked by reading, not compiled, not run on hardware
//
#include <hip/hip_runtime.h>
#include <hip/hip_bf16.h>

typedef __attribute__((ext_vector_type(16))) __bf16 v16bf;
typedef __attribute__((ext_vector_type(8)))  float  v8f;

#define N_NODES 50000
#define N_EDGES 800000
#define M_TILES 3125   // 50000 / 16

// ---------------------------------------------------------------------------
// Degree accumulation + inverse
// ---------------------------------------------------------------------------
__global__ void deg_accum_kernel(const long long* __restrict__ ei, float* __restrict__ deg) {
    int i = blockIdx.x * blockDim.x + threadIdx.x;
    if (i < N_EDGES) {
        int dst = (int)ei[N_EDGES + i];
        atomicAdd(&deg[dst], 1.0f);
    }
}

__global__ void deg_inv_kernel(float* __restrict__ deg) {
    int i = blockIdx.x * blockDim.x + threadIdx.x;
    if (i < N_NODES) deg[i] = 1.0f / fmaxf(deg[i], 1.0f);
}

// ---------------------------------------------------------------------------
// Edge scatter: agg[dst] += h[src]   (one block per edge; atomics resolve in L2)
// ---------------------------------------------------------------------------
__global__ void scatter_add_kernel(const float* __restrict__ hin,
                                   const long long* __restrict__ ei,
                                   float* __restrict__ agg, int din) {
    int e = blockIdx.x;
    int src = (int)ei[e];
    int dst = (int)ei[N_EDGES + e];
    const float* s = hin + (size_t)src * din;
    float*       d = agg + (size_t)dst * din;
    for (int f = threadIdx.x; f < din; f += blockDim.x)
        atomicAdd(&d[f], s[f]);
}

// ---------------------------------------------------------------------------
// Pack f32 row-major weight W[din x dout] into bf16, swizzled to the exact
// per-lane V_WMMA_*_16X16X32_BF16 B-matrix layout used here (contiguous k per lane half):
//   out[((kT*nTilesN + nT)*32 + lane)*16 + e]
//     holds W[k][n] with k = kT*32 + (lane>>4)*16 + e, n = nT*16 + (lane&15)
// so the GEMM loads B as one aligned 32-byte vector per lane per K-step.
// ---------------------------------------------------------------------------
__global__ void pack_weight_bf16_kernel(const float* __restrict__ W,
                                        __bf16* __restrict__ out,
                                        int din, int dout) {
    int idx = blockIdx.x * blockDim.x + threadIdx.x;
    int total = din * dout;
    if (idx >= total) return;
    int nTilesN = dout >> 4;
    int e    = idx & 15;
    int lane = (idx >> 4) & 31;
    int rest = idx >> 9;
    int nT   = rest % nTilesN;
    int kT   = rest / nTilesN;
    int k = kT * 32 + (lane >> 4) * 16 + e;
    int n = nT * 16 + (lane & 15);
    out[idx] = (__bf16)W[(size_t)k * dout + n];
}

// ---------------------------------------------------------------------------
// Fused dual-GEMM with bf16 WMMA (compile-time dims -> immediate offsets):
//   hout = (agg * invdeg) @ Wl + hin @ Wr + bl
// One wave32 computes one 16x16 output tile. B comes pre-swizzled in bf16.
// ---------------------------------------------------------------------------
template <int DIN, int DOUT>
__global__ void sage_gemm_wmma_kernel(const float* __restrict__ agg,
                                      const float* __restrict__ hin,
                                      const float* __restrict__ invdeg,
                                      const __bf16* __restrict__ BlSw,
                                      const __bf16* __restrict__ BrSw,
                                      const float* __restrict__ bl,
                                      float* __restrict__ hout) {
    constexpr int nTilesN    = DOUT / 16;
    constexpr int totalTiles = M_TILES * nTilesN;
    constexpr int BSTEP      = nTilesN * 32 * 16;   // bf16 elems per K-tile slab

    const int wave = blockIdx.x * (blockDim.x >> 5) + (threadIdx.x >> 5);
    if (wave >= totalTiles) return;          // wave-uniform: EXEC stays all-1s

    const int nT = wave % nTilesN;
    const int mT = wave / nTilesN;

    const int lane    = threadIdx.x & 31;
    const int l16     = lane & 15;
    const int halfSel = lane >> 4;           // 0: lanes 0-15, 1: lanes 16-31
    const int m       = mT * 16 + l16;       // A row held by this lane
    const int n       = nT * 16 + l16;       // C/D column held by this lane
    const int khA     = halfSel * 16;        // A-matrix K sub-offset (contiguous k, same as packed B)

    v8f acc = {};

    // ---- pass 1: A = agg * invdeg[m], B = packed Wl ----
    {
        const float scale = invdeg[m];
        const float*  Arow = agg + (size_t)m * DIN;
        const __bf16* Bp   = BlSw + ((size_t)nT * 32 + lane) * 16;
        for (int k0 = 0; k0 < DIN; k0 += 32, Bp += BSTEP) {
            v16bf va;
#pragma unroll
            for (int e = 0; e < 8; ++e) {
                va[e]     = (__bf16)(Arow[k0 + khA + e] * scale);
                va[e + 8] = (__bf16)(Arow[k0 + khA + 8 + e] * scale);
            }
            v16bf vb = *(const v16bf*)Bp;                 // 2x global_load_b128
            if (k0 + 32 < DIN) __builtin_prefetch(Bp + BSTEP, 0, 1);
            acc = __builtin_amdgcn_wmma_f32_16x16x32_bf16(
                      false, va, false, vb, (short)0, acc, false, false);
        }
    }
    // ---- pass 2: A = hin, B = packed Wr ----
    {
        const float*  Hrow = hin + (size_t)m * DIN;
        const __bf16* Bp   = BrSw + ((size_t)nT * 32 + lane) * 16;
        for (int k0 = 0; k0 < DIN; k0 += 32, Bp += BSTEP) {
            v16bf va;
#pragma unroll
            for (int e = 0; e < 8; ++e) {
                va[e]     = (__bf16)(Hrow[k0 + khA + e]);
                va[e + 8] = (__bf16)(Hrow[k0 + khA + 8 + e]);
            }
            v16bf vb = *(const v16bf*)Bp;
            if (k0 + 32 < DIN) __builtin_prefetch(Bp + BSTEP, 0, 1);
            acc = __builtin_amdgcn_wmma_f32_16x16x32_bf16(
                      false, va, false, vb, (short)0, acc, false, false);
        }
    }

    const float bias = bl[n];
#pragma unroll
    for (int r = 0; r < 8; ++r) {
        const int row = mT * 16 + r + 8 * halfSel;      // C/D layout (ISA 7.12.2)
        hout[(size_t)row * DOUT + n] = acc[r] + bias;
    }
}

// ---------------------------------------------------------------------------
// BatchNorm statistics + apply + ReLU
// ---------------------------------------------------------------------------
__global__ void col_stats_accum_kernel(const float* __restrict__ h,
                                       float* __restrict__ sums,
                                       float* __restrict__ sqs, int dout) {
    float s0 = 0.f, q0 = 0.f, s1 = 0.f, q1 = 0.f;   // dout <= 512 -> <=2 cols/thread
    const int c0 = threadIdx.x;
    const int c1 = threadIdx.x + 256;
    for (int r = blockIdx.x; r < N_NODES; r += gridDim.x) {
        const float* row = h + (size_t)r * dout;
        if (c0 < dout) { float v = row[c0]; s0 += v; q0 += v * v; }
        if (c1 < dout) { float v = row[c1]; s1 += v; q1 += v * v; }
    }
    if (c0 < dout) { atomicAdd(&sums[c0], s0); atomicAdd(&sqs[c0], q0); }
    if (c1 < dout) { atomicAdd(&sums[c1], s1); atomicAdd(&sqs[c1], q1); }
}

__global__ void col_stats_finalize_kernel(const float* __restrict__ sums,
                                          const float* __restrict__ sqs,
                                          float* __restrict__ meanb,
                                          float* __restrict__ rstdb, int dout) {
    int c = blockIdx.x * blockDim.x + threadIdx.x;
    if (c < dout) {
        const float invN = 1.0f / (float)N_NODES;
        float m = sums[c] * invN;
        float v = sqs[c] * invN - m * m;
        meanb[c] = m;
        rstdb[c] = rsqrtf(v + 1e-5f);
    }
}

__global__ void bn_relu_kernel(float* __restrict__ h,
                               const float* __restrict__ meanb,
                               const float* __restrict__ rstdb,
                               const float* __restrict__ g,
                               const float* __restrict__ b,
                               int dout, int total) {
    int idx = blockIdx.x * blockDim.x + threadIdx.x;
    if (idx >= total) return;
    int c = idx & (dout - 1);                 // dout is a power of two
    float v = (h[idx] - meanb[c]) * rstdb[c] * g[c] + b[c];
    h[idx] = fmaxf(v, 0.0f);
}

// ---------------------------------------------------------------------------
extern "C" void kernel_launch(void* const* d_in, const int* in_sizes, int n_in,
                              void* d_out, int out_size, void* d_ws, size_t ws_size,
                              hipStream_t stream) {
    const float*     x  = (const float*)d_in[0];
    const long long* ei = (const long long*)d_in[1];
    const float* Wl[3] = { (const float*)d_in[2],  (const float*)d_in[7],  (const float*)d_in[12] };
    const float* bl[3] = { (const float*)d_in[3],  (const float*)d_in[8],  (const float*)d_in[13] };
    const float* Wr[3] = { (const float*)d_in[4],  (const float*)d_in[9],  (const float*)d_in[14] };
    const float* gg[3] = { (const float*)d_in[5],  (const float*)d_in[10], (const float*)d_in[15] };
    const float* bb[3] = { (const float*)d_in[6],  (const float*)d_in[11], (const float*)d_in[16] };
    float* out = (float*)d_out;

    // ---- workspace layout (float units; all sub-buffers 32B aligned) ----
    float* ws      = (float*)d_ws;
    float* invdeg  = ws;                                   // 50048 (padded)
    float* h1      = invdeg + 50048;                       // 50000*512
    float* h2      = h1 + (size_t)N_NODES * 512;           // 50000*256
    float* agg     = h2 + (size_t)N_NODES * 256;           // 50000*512
    float* sums    = agg + (size_t)N_NODES * 512;          // 512
    float* sqs     = sums + 512;                           // 512
    float* meanb   = sqs + 512;                            // 512
    float* rstdb   = meanb + 512;                          // 512
    // bf16 packed weights (0.92 MB total)
    __bf16* wsw = (__bf16*)(rstdb + 512);
    const int wsz[3] = { 128 * 512, 512 * 256, 256 * 128 };
    __bf16* WlSw[3], * WrSw[3];
    {
        __bf16* p = wsw;
        for (int l = 0; l < 3; ++l) { WlSw[l] = p; p += wsz[l]; WrSw[l] = p; p += wsz[l]; }
    }

    // ---- degrees -> 1/max(deg,1) ----
    hipMemsetAsync(invdeg, 0, (size_t)N_NODES * sizeof(float), stream);
    deg_accum_kernel<<<(N_EDGES + 255) / 256, 256, 0, stream>>>(ei, invdeg);
    deg_inv_kernel<<<(N_NODES + 255) / 256, 256, 0, stream>>>(invdeg);

    // ---- pack all weights to swizzled bf16 (tiny, reused 3125x per tile-col) ----
    const int dims[4] = { 128, 512, 256, 128 };
    for (int l = 0; l < 3; ++l) {
        int total = wsz[l];
        pack_weight_bf16_kernel<<<(total + 255) / 256, 256, 0, stream>>>(
            Wl[l], WlSw[l], dims[l], dims[l + 1]);
        pack_weight_bf16_kernel<<<(total + 255) / 256, 256, 0, stream>>>(
            Wr[l], WrSw[l], dims[l], dims[l + 1]);
    }

    const float* hin = x;
    for (int l = 0; l < 3; ++l) {
        const int din  = dims[l];
        const int dout = dims[l + 1];
        float* hout = (l == 0) ? h1 : (l == 1) ? h2 : out;

        // mean aggregation (sum part; 1/deg folded into GEMM A-load)
        hipMemsetAsync(agg, 0, (size_t)N_NODES * din * sizeof(float), stream);
        scatter_add_kernel<<<N_EDGES, 128, 0, stream>>>(hin, ei, agg, din);

        // fused dual GEMM + bias via bf16 WMMA (8 waves / 256-thread block)
        const int nTilesN    = dout / 16;
        const int totalTiles = M_TILES * nTilesN;
        const int blocks     = (totalTiles + 7) / 8;
        if (l == 0)
            sage_gemm_wmma_kernel<128, 512><<<blocks, 256, 0, stream>>>(
                agg, hin, invdeg, WlSw[0], WrSw[0], bl[0], hout);
        else if (l == 1)
            sage_gemm_wmma_kernel<512, 256><<<blocks, 256, 0, stream>>>(
                agg, hin, invdeg, WlSw[1], WrSw[1], bl[1], hout);
        else
            sage_gemm_wmma_kernel<256, 128><<<blocks, 256, 0, stream>>>(
                agg, hin, invdeg, WlSw[2], WrSw[2], bl[2], hout);

        // batch-norm stats + apply + relu
        hipMemsetAsync(sums, 0, dout * sizeof(float), stream);
        hipMemsetAsync(sqs,  0, dout * sizeof(float), stream);
        col_stats_accum_kernel<<<256, 256, 0, stream>>>(hout, sums, sqs, dout);
        col_stats_finalize_kernel<<<(dout + 255) / 256, 256, 0, stream>>>(sums, sqs, meanb, rstdb, dout);
        const int total = N_NODES * dout;
        bn_relu_kernel<<<(total + 255) / 256, 256, 0, stream>>>(hout, meanb, rstdb, gg[l], bb[l], dout, total);

        hin = hout;
    }
}
